// RelPosMultiHeadSelfAttention_15324443312701
// MI455X (gfx1250) — matrix operand order repaired and checked by reading, not compiled, not run on hardware
//
#include <hip/hip_runtime.h>
#include <math.h>

typedef __bf16 bf16_t;
typedef __attribute__((ext_vector_type(16))) __bf16 v16bf;
typedef __attribute__((ext_vector_type(8)))  __bf16 v8bf;
typedef __attribute__((ext_vector_type(8)))  float  v8f;

union FragU { v16bf v; v8bf h[2]; };

// ---------------------------------------------------------------------------
// Async global -> LDS copy, 16B per lane, tracked by ASYNCcnt (CDNA5 TDM-lite
// path).  VDST = per-lane LDS byte offset (low 32 bits of a generic LDS
// pointer).
// ---------------------------------------------------------------------------
__device__ __forceinline__ void async_g2l_b128(unsigned lds_off, const void* gptr) {
  unsigned long long ga = (unsigned long long)(uintptr_t)gptr;
  asm volatile("global_load_async_to_lds_b128 %0, %1, off"
               :: "v"(lds_off), "v"(ga) : "memory");
}
__device__ __forceinline__ void wait_async0() {
  asm volatile("s_wait_asynccnt 0" ::: "memory");
}

// ---------------------------------------------------------------------------
// WMMA fragment helpers (wave32, 16x16x32 bf16, f32 accum)
// A (16x32, MxK): lane m=lane&15, g=lane>>4; elem j -> k = (j>>3)*16 + g*8 + (j&7)
//   => two contiguous 8-elem (16B) runs per lane at k = g*8 and 16+g*8.
// B (32x16, KxN): lane n=lane&15, g=lane>>4; elem j -> k = (j>>3)*16 + g*8 + (j&7)
//   => two contiguous 8-elem runs per lane; we store B transposed (N-major).
// C/D (16x16 f32): vgpr v, lane n=lane&15, g=lane>>4 -> element (m=v+8g, n).
// ---------------------------------------------------------------------------
__device__ __forceinline__ v16bf load_frag_a(const bf16_t* base, int ld) {
  const int lane = threadIdx.x & 31;
  const int m = lane & 15, g = lane >> 4;
  const bf16_t* p = base + m * ld + g * 8;
  FragU f;
  f.h[0] = *(const v8bf*)(p);
  f.h[1] = *(const v8bf*)(p + 16);
  return f.v;
}

__device__ __forceinline__ v16bf load_frag_b(const bf16_t* baseT, int ld) {
  const int lane = threadIdx.x & 31;
  const int n = lane & 15, g = lane >> 4;
  const bf16_t* p = baseT + n * ld + g * 8;
  FragU f;
  f.h[0] = *(const v8bf*)(p);
  f.h[1] = *(const v8bf*)(p + 16);
  return f.v;
}

__device__ __forceinline__ v8f wmma_bf16(v16bf a, v16bf b, v8f c) {
  return __builtin_amdgcn_wmma_f32_16x16x32_bf16(false, a, false, b, (short)0, c,
                                                 false, false);
}

// ---------------------------------------------------------------------------
// Elementwise prep kernels
// ---------------------------------------------------------------------------
__global__ void cvt_bf16_kernel(const float* __restrict__ src, bf16_t* __restrict__ dst, int n) {
  int i = blockIdx.x * blockDim.x + threadIdx.x;
  if (i < n) dst[i] = (bf16_t)src[i];
}

// Wt[c][r] = W[r][c]  (512x512), bf16 output (N-major for B fragments)
__global__ void transpose_w_kernel(const float* __restrict__ W, bf16_t* __restrict__ Wt) {
  int i = blockIdx.x * blockDim.x + threadIdx.x;
  if (i >= 512 * 512) return;
  int r = i >> 9, c = i & 511;
  Wt[(size_t)c * 512 + r] = (bf16_t)W[i];
}

// Sinusoidal relative PE, 2047 x 512 (+1 zero pad row), bf16 (GEMM A input)
__global__ void build_pe_kernel(bf16_t* __restrict__ pe) {
  int i = blockIdx.x * blockDim.x + threadIdx.x;
  if (i >= 2048 * 512) return;
  int p = i >> 9, col = i & 511;
  if (p >= 2047) { pe[i] = (bf16_t)0.0f; return; }   // pad row: keep async GEMM read clean
  float pos = 1023.0f - (float)p;
  float e = (2.0f * (float)(col >> 1) / 512.0f) * 9.210340371976184f; // ln(10000)
  float inv = __expf(e);
  float ang = pos / inv;
  pe[i] = (bf16_t)((col & 1) ? cosf(ang) : sinf(ang));
}

// ---------------------------------------------------------------------------
// Tiled bf16 WMMA GEMM:  C(MxN f32) = A(MxK bf16) * Bt^T + bias,  N=K=512
// Bt is N-major (N x K).  Block tile 128x64, 8 waves, wave w owns rows 16w..16w+15.
// Tiles staged with global_load_async_to_lds_b128 (ASYNCcnt).  A rows past M
// may stage garbage (still-allocated pad rows); their accumulator rows are
// discarded by the guarded epilogue store.
// ---------------------------------------------------------------------------
__global__ __launch_bounds__(256)
void gemm_bf16_kernel(const bf16_t* __restrict__ A, const bf16_t* __restrict__ Bt,
                      const float* __restrict__ bias, float* __restrict__ C, int M) {
  __shared__ bf16_t sA[128][32];
  __shared__ bf16_t sB[64][32];
  const int tid = threadIdx.x;
  const int wave = tid >> 5;
  const int lane = tid & 31;
  const int n_ = lane & 15, g = lane >> 4;
  const int m0 = blockIdx.x * 128;
  const int n0 = blockIdx.y * 64;
  v8f acc[4] = {};

  const int ra = tid >> 1, sa = tid & 1;       // A: row, 16-elem seg
  const int rb = tid >> 2, sb = tid & 3;       // B: row, 8-elem seg
  const bf16_t* srcA = A + (size_t)(m0 + ra) * 512 + sa * 16;
  const bf16_t* srcB = Bt + (size_t)(n0 + rb) * 512 + sb * 8;
  const unsigned dstA = (unsigned)(uintptr_t)&sA[ra][sa * 16];
  const unsigned dstB = (unsigned)(uintptr_t)&sB[rb][sb * 8];

  for (int k0 = 0; k0 < 512; k0 += 32) {
    __syncthreads();
    async_g2l_b128(dstA,      srcA + k0);
    async_g2l_b128(dstA + 16, srcA + k0 + 8);
    async_g2l_b128(dstB,      srcB + k0);
    wait_async0();
    __syncthreads();

    v16bf a = load_frag_a(&sA[wave * 16][0], 32);
#pragma unroll
    for (int nc = 0; nc < 4; ++nc) {
      v16bf b = load_frag_b(&sB[nc * 16][0], 32);
      acc[nc] = wmma_bf16(a, b, acc[nc]);
    }
  }

#pragma unroll
  for (int nc = 0; nc < 4; ++nc) {
    int col = n0 + nc * 16 + n_;
    float bb = bias[col];
#pragma unroll
    for (int v = 0; v < 8; ++v) {
      int row = m0 + wave * 16 + v + 8 * g;
      if (row < M) C[(size_t)row * 512 + col] = acc[nc][v] + bb;
    }
  }
}

// ---------------------------------------------------------------------------
// Pack projections into head-major bf16 layouts
//   qu/qv/kh: [bh][t][64],  vT: [bh][64][t]
// ---------------------------------------------------------------------------
__global__ void pack_qkv_kernel(const float* __restrict__ qf, const float* __restrict__ kf,
                                const float* __restrict__ vf, const float* __restrict__ u,
                                const float* __restrict__ vv, bf16_t* __restrict__ qu,
                                bf16_t* __restrict__ qv, bf16_t* __restrict__ kh,
                                bf16_t* __restrict__ vt) {
  int i = blockIdx.x * blockDim.x + threadIdx.x;
  if (i >= 4096 * 512) return;
  int col = i & 511, row = i >> 9;           // row = b*1024 + t
  int head = col >> 6, dd = col & 63;
  int b = row >> 10, t = row & 1023;
  int bh = b * 8 + head;
  size_t hd = ((size_t)bh * 1024 + t) * 64 + dd;
  float qq = qf[i];
  qu[hd] = (bf16_t)(qq + u[col]);
  qv[hd] = (bf16_t)(qq + vv[col]);
  kh[hd] = (bf16_t)kf[i];
  vt[((size_t)bh * 64 + dd) * 1024 + t] = (bf16_t)vf[i];
}

// E f32 (2047 x 512) -> per-head bf16 [h][2048][64], padded row 2047 = 0
__global__ void pack_E_kernel(const float* __restrict__ Ef, bf16_t* __restrict__ Eb) {
  int i = blockIdx.x * blockDim.x + threadIdx.x;
  if (i >= 2048 * 512) return;
  int col = i & 511, l = i >> 9;
  int head = col >> 6, dd = col & 63;
  float val = (l < 2047) ? Ef[(size_t)l * 512 + col] : 0.0f;
  Eb[((size_t)head * 2048 + l) * 64 + dd] = (bf16_t)val;
}

// ---------------------------------------------------------------------------
// Fused relative-position flash attention.
//   grid = (T/128, B*H), block = 256 (8 waves); wave owns 16 query rows.
//   scores[t,s] = (qu[t]·k[s] + qv[t]·E[1023+s-t]) / 8, softmax over s, *V.
// E-term per 16x16 tile: M[r,j] = qv[t0+r]·E[Lbase-15+j] (16x32 via WMMA),
// then diagonal gather M[r, c-r+15] through per-wave LDS.
// K/V tiles staged via global_load_async_to_lds_b128 (ASYNCcnt).
// ---------------------------------------------------------------------------
__global__ __launch_bounds__(256)
void attn_kernel(const bf16_t* __restrict__ QU, const bf16_t* __restrict__ QV,
                 const bf16_t* __restrict__ KH, const bf16_t* __restrict__ VT,
                 const bf16_t* __restrict__ EB, bf16_t* __restrict__ Oout) {
  __shared__ bf16_t sK[128][64];      // keys, [s][dh]
  __shared__ bf16_t sV[64][128];      // values transposed, [dh][s]
  __shared__ float  sM[8][16][33];    // per-wave E-band product
  __shared__ bf16_t sP[8][16][32];    // per-wave probabilities

  const int tid = threadIdx.x;
  const int wave = tid >> 5;
  const int lane = tid & 31;
  const int n_ = lane & 15, g = lane >> 4;
  const int bh = blockIdx.y;
  const int head = bh & 7;
  const int b = bh >> 3;
  const int t0 = blockIdx.x * 128 + wave * 16;

  const bf16_t* quB = QU + ((size_t)bh * 1024 + t0) * 64;
  const bf16_t* qvB = QV + ((size_t)bh * 1024 + t0) * 64;
  v16bf qu0 = load_frag_a(quB, 64);
  v16bf qu1 = load_frag_a(quB + 32, 64);
  v16bf qv0 = load_frag_a(qvB, 64);
  v16bf qv1 = load_frag_a(qvB + 32, 64);
  const bf16_t* Eh = EB + (size_t)head * 2048 * 64;

  // Per-thread staging assignments (64B each tile)
  const int rk = tid >> 1, hk = tid & 1;
  const int rv = tid >> 2, qd = tid & 3;
  const bf16_t* srcK = KH + ((size_t)bh * 1024 + rk) * 64 + hk * 32;
  const bf16_t* srcV = VT + ((size_t)bh * 64 + rv) * 1024 + qd * 32;
  const unsigned dstK = (unsigned)(uintptr_t)&sK[rk][hk * 32];
  const unsigned dstV = (unsigned)(uintptr_t)&sV[rv][qd * 32];

  v8f o[4] = {};
  float mrow[8], lrow[8];
#pragma unroll
  for (int v = 0; v < 8; ++v) { mrow[v] = -3.0e38f; lrow[v] = 0.0f; }

  for (int s0 = 0; s0 < 1024; s0 += 128) {
    __syncthreads();
#pragma unroll
    for (int i = 0; i < 4; ++i) {           // 64B per thread per tile
      async_g2l_b128(dstK + i * 16, srcK + (size_t)s0 * 64 + i * 8);
      async_g2l_b128(dstV + i * 16, srcV + s0 + i * 8);
    }
    if (s0 + 128 < 1024) // hint next K tile toward L2 (global_prefetch_b8)
      __builtin_prefetch(KH + ((size_t)bh * 1024 + s0 + 128 + rk) * 64, 0, 1);
    wait_async0();
    __syncthreads();

    for (int st = 0; st < 4; ++st) {
      float sv[2][8];
#pragma unroll
      for (int su = 0; su < 2; ++su) {
        const int srel = st * 32 + su * 16;
        const int sg = s0 + srel;
        // QK^T tile
        v8f sk = {};
        sk = wmma_bf16(qu0, load_frag_b(&sK[srel][0], 64), sk);
        sk = wmma_bf16(qu1, load_frag_b(&sK[srel][32], 64), sk);
        // E band: rows Lbase-15 .. Lbase+16, Lbase = 1023 + sg - t0
        const bf16_t* Ebase = Eh + (size_t)(1023 + sg - t0 - 15) * 64;
        v8f me0 = {}, me1 = {};
        me0 = wmma_bf16(qv0, load_frag_b(Ebase, 64), me0);
        me0 = wmma_bf16(qv1, load_frag_b(Ebase + 32, 64), me0);
        me1 = wmma_bf16(qv0, load_frag_b(Ebase + 16 * 64, 64), me1);
        me1 = wmma_bf16(qv1, load_frag_b(Ebase + 16 * 64 + 32, 64), me1);
#pragma unroll
        for (int v = 0; v < 8; ++v) {
          sM[wave][v + 8 * g][n_]      = me0[v];
          sM[wave][v + 8 * g][16 + n_] = me1[v];
        }
        asm volatile("s_wait_dscnt 0" ::: "memory");
#pragma unroll
        for (int v = 0; v < 8; ++v) {
          int r = v + 8 * g;
          sv[su][v] = (sk[v] + sM[wave][r][n_ - r + 15]) * 0.125f;
        }
      }
      // Online softmax update over this 16x32 chunk
      float scl[8];
#pragma unroll
      for (int v = 0; v < 8; ++v) {
        float mx = fmaxf(sv[0][v], sv[1][v]);
#pragma unroll
        for (int d = 1; d < 16; d <<= 1) mx = fmaxf(mx, __shfl_xor(mx, d, 32));
        float mnew = fmaxf(mrow[v], mx);
        float sc = __expf(mrow[v] - mnew);
        float p0 = __expf(sv[0][v] - mnew);
        float p1 = __expf(sv[1][v] - mnew);
        float rs = p0 + p1;
#pragma unroll
        for (int d = 1; d < 16; d <<= 1) rs += __shfl_xor(rs, d, 32);
        lrow[v] = lrow[v] * sc + rs;
        mrow[v] = mnew;
        scl[v] = sc;
        sP[wave][v + 8 * g][n_]      = (bf16_t)p0;
        sP[wave][v + 8 * g][16 + n_] = (bf16_t)p1;
      }
      asm volatile("s_wait_dscnt 0" ::: "memory");
      v16bf pa = load_frag_a(&sP[wave][0][0], 32);
#pragma unroll
      for (int nc = 0; nc < 4; ++nc) {
#pragma unroll
        for (int v = 0; v < 8; ++v) o[nc][v] *= scl[v];
        v16bf bv = load_frag_b(&sV[nc * 16][st * 32], 128);
        o[nc] = wmma_bf16(pa, bv, o[nc]);
      }
    }
  }

  // Epilogue: normalize and store merged heads [b][t][h*64+dh] as bf16
#pragma unroll
  for (int nc = 0; nc < 4; ++nc) {
#pragma unroll
    for (int v = 0; v < 8; ++v) {
      int r = v + 8 * g;
      int t = t0 + r;
      float val = o[nc][v] / lrow[v];
      Oout[((size_t)(b * 1024 + t)) * 512 + head * 64 + nc * 16 + n_] = (bf16_t)val;
    }
  }
}

// ---------------------------------------------------------------------------
extern "C" void kernel_launch(void* const* d_in, const int* in_sizes, int n_in,
                              void* d_out, int out_size, void* d_ws, size_t ws_size,
                              hipStream_t stream) {
  const float* Q  = (const float*)d_in[0];
  const float* K  = (const float*)d_in[1];
  const float* V  = (const float*)d_in[2];
  const float* Wq = (const float*)d_in[3];
  const float* bq = (const float*)d_in[4];
  const float* Wk = (const float*)d_in[5];
  const float* bk = (const float*)d_in[6];
  const float* Wv = (const float*)d_in[7];
  const float* bv = (const float*)d_in[8];
  const float* Wp = (const float*)d_in[9];
  const float* bp = (const float*)d_in[10];
  const float* Wo = (const float*)d_in[11];
  const float* bo = (const float*)d_in[12];
  const float* u  = (const float*)d_in[13];
  const float* vV = (const float*)d_in[14];

  char* ws = (char*)d_ws;
  size_t off = 0;
  auto alloc = [&](size_t bytes) -> void* {
    void* p = ws + off;
    off += (bytes + 255) & ~(size_t)255;
    return p;
  };

  const size_t MD = (size_t)4096 * 512;
  bf16_t* Qbf  = (bf16_t*)alloc(MD * 2);
  bf16_t* Kbf  = (bf16_t*)alloc(MD * 2);
  bf16_t* Vbf  = (bf16_t*)alloc(MD * 2);
  bf16_t* WqT  = (bf16_t*)alloc(512 * 512 * 2);
  bf16_t* WkT  = (bf16_t*)alloc(512 * 512 * 2);
  bf16_t* WvT  = (bf16_t*)alloc(512 * 512 * 2);
  bf16_t* WpT  = (bf16_t*)alloc(512 * 512 * 2);
  bf16_t* WoT  = (bf16_t*)alloc(512 * 512 * 2);
  bf16_t* pebf = (bf16_t*)alloc((size_t)2048 * 512 * 2);
  float*  qf   = (float*)alloc(MD * 4);
  float*  kf   = (float*)alloc(MD * 4);
  float*  vf   = (float*)alloc(MD * 4);
  float*  Ef   = (float*)alloc((size_t)2048 * 512 * 4);
  bf16_t* qu   = (bf16_t*)alloc(MD * 2);
  bf16_t* qv   = (bf16_t*)alloc(MD * 2);
  bf16_t* kh   = (bf16_t*)alloc(MD * 2);
  bf16_t* vT   = (bf16_t*)alloc(MD * 2);
  bf16_t* Ebf  = (bf16_t*)alloc((size_t)8 * 2048 * 64 * 2);
  bf16_t* attn = (bf16_t*)alloc(MD * 2);
  (void)ws_size; (void)in_sizes; (void)n_in; (void)out_size;

  const int TPB = 256;
  // --- prep ---
  cvt_bf16_kernel<<<(int)((MD + TPB - 1) / TPB), TPB, 0, stream>>>(Q, Qbf, (int)MD);
  cvt_bf16_kernel<<<(int)((MD + TPB - 1) / TPB), TPB, 0, stream>>>(K, Kbf, (int)MD);
  cvt_bf16_kernel<<<(int)((MD + TPB - 1) / TPB), TPB, 0, stream>>>(V, Vbf, (int)MD);
  transpose_w_kernel<<<(512 * 512 + TPB - 1) / TPB, TPB, 0, stream>>>(Wq, WqT);
  transpose_w_kernel<<<(512 * 512 + TPB - 1) / TPB, TPB, 0, stream>>>(Wk, WkT);
  transpose_w_kernel<<<(512 * 512 + TPB - 1) / TPB, TPB, 0, stream>>>(Wv, WvT);
  transpose_w_kernel<<<(512 * 512 + TPB - 1) / TPB, TPB, 0, stream>>>(Wp, WpT);
  transpose_w_kernel<<<(512 * 512 + TPB - 1) / TPB, TPB, 0, stream>>>(Wo, WoT);
  build_pe_kernel<<<(2048 * 512 + TPB - 1) / TPB, TPB, 0, stream>>>(pebf);

  // --- projection GEMMs (WMMA) ---
  gemm_bf16_kernel<<<dim3(32, 8), TPB, 0, stream>>>(Qbf, WqT, bq, qf, 4096);
  gemm_bf16_kernel<<<dim3(32, 8), TPB, 0, stream>>>(Kbf, WkT, bk, kf, 4096);
  gemm_bf16_kernel<<<dim3(32, 8), TPB, 0, stream>>>(Vbf, WvT, bv, vf, 4096);
  gemm_bf16_kernel<<<dim3(16, 8), TPB, 0, stream>>>(pebf, WpT, bp, Ef, 2047);

  // --- pack to head-major bf16 ---
  pack_qkv_kernel<<<(int)((MD + TPB - 1) / TPB), TPB, 0, stream>>>(qf, kf, vf, u, vV,
                                                                   qu, qv, kh, vT);
  pack_E_kernel<<<(2048 * 512 + TPB - 1) / TPB, TPB, 0, stream>>>(Ef, Ebf);

  // --- fused flash attention with relative-position band (WMMA) ---
  attn_kernel<<<dim3(8, 32), TPB, 0, stream>>>(qu, qv, kh, vT, Ebf, attn);

  // --- output projection (WMMA), f32 result ---
  gemm_bf16_kernel<<<dim3(32, 8), TPB, 0, stream>>>(attn, WoT, bo, (float*)d_out, 4096);
}
